// SlidingWindowAttention_17721035063824
// MI455X (gfx1250) — matrix operand order repaired and checked by reading, not compiled, not run on hardware
//
#include <hip/hip_runtime.h>

#define NHEADS 16
#define HD     64
#define SEQ    2048
#define BATCH  2
#define MROWS  4096   // B*S
#define HID    1024
#define WIN    256

typedef __attribute__((ext_vector_type(16))) __bf16 v16bf;
typedef __attribute__((ext_vector_type(8)))  float  v8f;

union ABfrag { v16bf v; uint4 q[2]; };

__device__ __forceinline__ unsigned short f2b(float f) {
  unsigned u = __float_as_uint(f);
  u += 0x7FFFu + ((u >> 16) & 1u);          // round-to-nearest-even
  return (unsigned short)(u >> 16);
}
__device__ __forceinline__ unsigned pack2(float a, float b) {
  return (unsigned)f2b(a) | ((unsigned)f2b(b) << 16);
}

__global__ void f32_to_bf16_kernel(const float* __restrict__ in,
                                   unsigned short* __restrict__ out, int n) {
  int i = blockIdx.x * blockDim.x + threadIdx.x;
  if (i < n) out[i] = f2b(in[i]);
}

#define MODE_QKNORM 0
#define MODE_VT     1
#define MODE_F32OUT 2

// C[m, head*64+d] = sum_k X[m,k] * W[head*64+d, k]   (i.e. x @ W^T)
// One wave per (16-row m-tile, head). 128 bf16 WMMAs per wave.
__global__ __launch_bounds__(256) void proj_kernel(
    const unsigned short* __restrict__ X,   // [4096,1024] bf16
    const unsigned short* __restrict__ W,   // [1024,1024] bf16
    const float* __restrict__ normw,        // [64] or unused
    void* __restrict__ outp, int mode) {
  const int lane = threadIdx.x & 31;
  const int wid  = threadIdx.x >> 5;
  const int gw   = blockIdx.x * 8 + wid;    // 0..4095
  const int mt   = gw & 255;
  const int head = gw >> 8;                 // 0..15
  const int n    = lane & 15;
  const int hi   = lane >> 4;
  const int m0   = mt * 16;

  const unsigned short* xrow = X + (size_t)(m0 + n) * HID;
  v8f acc[4] = {};
  for (int kb = 0; kb < HID; kb += 32) {
    ABfrag a;   // A-layout: lanes0-15 hold K kb..kb+15, lanes16-31 kb+16..kb+31 (contiguous)
    a.q[0] = *(const uint4*)(xrow + kb + (hi ? 16 : 0));
    a.q[1] = *(const uint4*)(xrow + kb + 8 + (hi ? 16 : 0));
#pragma unroll
    for (int j = 0; j < 4; ++j) {
      const unsigned short* wrow =
          W + (size_t)(head * 64 + j * 16 + n) * HID + kb + (hi ? 16 : 0);
      ABfrag b;  // B-layout: lane=column, lanes0-15 k=0..15, lanes16-31 k=16..31 (contiguous)
      b.q[0] = *(const uint4*)(wrow);
      b.q[1] = *(const uint4*)(wrow + 8);
      acc[j] = __builtin_amdgcn_wmma_f32_16x16x32_bf16(
          false, a.v, false, b.v, (short)0, acc[j], false, false);
    }
  }

  const int b  = m0 >> 11;            // tile never crosses batch boundary
  const int s0 = m0 & (SEQ - 1);

  if (mode == MODE_QKNORM) {
    // RMSNorm over head_dim: row m = r + 8*hi lives across 16 lanes x 4 accs
    float ss[8];
#pragma unroll
    for (int r = 0; r < 8; ++r) {
      float t = 0.f;
#pragma unroll
      for (int j = 0; j < 4; ++j) t += acc[j][r] * acc[j][r];
      ss[r] = t;
    }
#pragma unroll
    for (int r = 0; r < 8; ++r) {
      ss[r] += __shfl_xor(ss[r], 1, 16);
      ss[r] += __shfl_xor(ss[r], 2, 16);
      ss[r] += __shfl_xor(ss[r], 4, 16);
      ss[r] += __shfl_xor(ss[r], 8, 16);
    }
    unsigned short* o = (unsigned short*)outp;
#pragma unroll
    for (int j = 0; j < 4; ++j) {
      const float wd = normw[j * 16 + n];
#pragma unroll
      for (int r = 0; r < 8; ++r) {
        float rms = rsqrtf(ss[r] * (1.f / 64.f) + 1e-6f);
        float v = acc[j][r] * rms * wd;
        int s = s0 + r + 8 * hi;
        o[((size_t)(b * NHEADS + head) * SEQ + s) * HD + j * 16 + n] = f2b(v);
      }
    }
  } else if (mode == MODE_VT) {
    unsigned short* o = (unsigned short*)outp;  // [b,h,d,s]
#pragma unroll
    for (int j = 0; j < 4; ++j)
#pragma unroll
      for (int r = 0; r < 8; ++r) {
        int s = s0 + r + 8 * hi;
        o[((size_t)(b * NHEADS + head) * HD + j * 16 + n) * SEQ + s] =
            f2b(acc[j][r]);
      }
  } else {  // MODE_F32OUT
    float* o = (float*)outp;
#pragma unroll
    for (int j = 0; j < 4; ++j)
#pragma unroll
      for (int r = 0; r < 8; ++r) {
        int m = m0 + r + 8 * hi;
        o[(size_t)m * HID + head * 64 + j * 16 + n] = acc[j][r];
      }
  }
}

// One wave per (b, h, 16-query tile). Scores computed transposed:
// S^T[j, q] = sum_d K[j,d] * Q[q,d]  ->  A=K tile (rows=keys), B=Q^T (lane=query col).
__global__ __launch_bounds__(256) void attn_kernel(
    const unsigned short* __restrict__ Qbf,  // [b,h,s,d] bf16
    const unsigned short* __restrict__ Kbf,  // [b,h,s,d] bf16
    const unsigned short* __restrict__ Vt,   // [b,h,d,s] bf16
    float* __restrict__ attn,                // [b,h,S,S] fp32 (pre-zeroed)
    unsigned short* __restrict__ ctx) {      // [b*S, 1024] bf16
  __shared__ __align__(16) unsigned short ptile[8][16 * 32];  // per-wave P staging

  const int lane = threadIdx.x & 31;
  const int wid  = threadIdx.x >> 5;
  const int gw   = blockIdx.x * 8 + wid;    // 0..4095
  const int qt   = gw & 127;
  const int bh   = gw >> 7;                 // b*16+h
  const int n    = lane & 15;
  const int hi   = lane >> 4;
  const int q0   = qt * 16;
  const float NEGINF = -__builtin_inff();
  const float SCALE  = 0.125f;              // 1/sqrt(64)

  const unsigned short* qbase = Qbf + (size_t)bh * SEQ * HD;
  const unsigned short* kbase = Kbf + (size_t)bh * SEQ * HD;
  const unsigned short* vbase = Vt + (size_t)bh * HD * SEQ;
  float* abase = attn + (size_t)bh * SEQ * SEQ;

  // Q as B operand (lane = query column, contiguous rows of Q)
  ABfrag bq0, bq1;
  {
    const unsigned short* qrow = qbase + (size_t)(q0 + n) * HD + (hi ? 8 : 0);
    bq0.q[0] = *(const uint4*)(qrow);
    bq0.q[1] = *(const uint4*)(qrow + 16);
    bq1.q[0] = *(const uint4*)(qrow + 32);
    bq1.q[1] = *(const uint4*)(qrow + 48);
  }

  const int jt0 = (qt >= 16) ? (qt - 16) : 0;

  // ---- sweep 1: exact softmax stats (online max/sum, no V) ----
  float mrun = NEGINF, lrun = 0.f;
  for (int kt = jt0; kt <= qt; ++kt) {
    const int j0 = kt * 16;
    const unsigned short* krow = kbase + (size_t)(j0 + n) * HD;
    ABfrag ak0, ak1;
    ak0.q[0] = *(const uint4*)(krow + (hi ? 8 : 0));
    ak0.q[1] = *(const uint4*)(krow + 16 + (hi ? 8 : 0));
    ak1.q[0] = *(const uint4*)(krow + 32 + (hi ? 8 : 0));
    ak1.q[1] = *(const uint4*)(krow + 48 + (hi ? 8 : 0));
    v8f s = {};
    s = __builtin_amdgcn_wmma_f32_16x16x32_bf16(false, ak0.v, false, bq0.v, (short)0, s, false, false);
    s = __builtin_amdgcn_wmma_f32_16x16x32_bf16(false, ak1.v, false, bq1.v, (short)0, s, false, false);
    float sv[8], tmax = NEGINF;
    const int i = q0 + n;
#pragma unroll
    for (int r = 0; r < 8; ++r) {
      int j = j0 + r + 8 * hi;
      bool valid = (j <= i) && (i - j < WIN);
      sv[r] = valid ? s[r] * SCALE : NEGINF;
      tmax = fmaxf(tmax, sv[r]);
    }
    float mnew = fmaxf(mrun, tmax);
    if (mnew != NEGINF) {
      float alpha = (mrun != NEGINF) ? __expf(mrun - mnew) : 0.f;
      float acc = 0.f;
#pragma unroll
      for (int r = 0; r < 8; ++r)
        acc += (sv[r] != NEGINF) ? __expf(sv[r] - mnew) : 0.f;
      lrun = lrun * alpha + acc;
      mrun = mnew;
    }
  }
  // combine the two wave halves (same query column n)
  const float mo = __shfl_xor(mrun, 16, 32);
  const float lo = __shfl_xor(lrun, 16, 32);
  const float mt2 = fmaxf(mrun, mo);
  float lt = 0.f;
  if (mrun != NEGINF) lt += lrun * __expf(mrun - mt2);
  if (mo   != NEGINF) lt += lo   * __expf(mo - mt2);
  const float inv = 1.f / lt;   // diagonal always valid -> lt > 0

  // ---- sweep 2: normalized probs -> d_out, P staged in LDS, P·V WMMA ----
  v8f o[4] = {};
  unsigned short* myP = &ptile[wid][0];
  for (int kp = jt0; kp <= qt; kp += 2) {
#pragma unroll
    for (int t = 0; t < 2; ++t) {
      const int kt = kp + t;
      if (kt <= qt) {
        const int j0 = kt * 16;
        const unsigned short* krow = kbase + (size_t)(j0 + n) * HD;
        ABfrag ak0, ak1;
        ak0.q[0] = *(const uint4*)(krow + (hi ? 8 : 0));
        ak0.q[1] = *(const uint4*)(krow + 16 + (hi ? 8 : 0));
        ak1.q[0] = *(const uint4*)(krow + 32 + (hi ? 8 : 0));
        ak1.q[1] = *(const uint4*)(krow + 48 + (hi ? 8 : 0));
        v8f s = {};
        s = __builtin_amdgcn_wmma_f32_16x16x32_bf16(false, ak0.v, false, bq0.v, (short)0, s, false, false);
        s = __builtin_amdgcn_wmma_f32_16x16x32_bf16(false, ak1.v, false, bq1.v, (short)0, s, false, false);
        float p[8];
        const int i = q0 + n;
#pragma unroll
        for (int r = 0; r < 8; ++r) {
          int j = j0 + r + 8 * hi;
          bool valid = (j <= i) && (i - j < WIN);
          p[r] = valid ? __expf(s[r] * SCALE - mt2) * inv : 0.f;
        }
        // contiguous fp32 attn rows: lane owns 8 consecutive j for query i
        float* arow = abase + (size_t)i * SEQ + j0 + hi * 8;
        *(float4*)(arow)     = make_float4(p[0], p[1], p[2], p[3]);
        *(float4*)(arow + 4) = make_float4(p[4], p[5], p[6], p[7]);
        // stage P (bf16) to LDS at [q=n][jcol], D-layout -> row-major
        uint4 pw;
        pw.x = pack2(p[0], p[1]); pw.y = pack2(p[2], p[3]);
        pw.z = pack2(p[4], p[5]); pw.w = pack2(p[6], p[7]);
        *(uint4*)&myP[(size_t)n * 32 + t * 16 + hi * 8] = pw;
      } else {
        *(uint4*)&myP[(size_t)n * 32 + 16 + hi * 8] = make_uint4(0, 0, 0, 0);
      }
    }
    asm volatile("s_wait_dscnt 0" ::: "memory");
    // read P back in A-layout (rows = queries, K = 32 keys)
    ABfrag ap;
    const unsigned short* prow = myP + (size_t)n * 32;
    ap.q[0] = *(const uint4*)(prow + (hi ? 16 : 0));
    ap.q[1] = *(const uint4*)(prow + 8 + (hi ? 16 : 0));
    asm volatile("" ::: "memory");
    // B = V^T slice: lane = d column, contiguous j run from pre-transposed V
    int jb = kp * 16 + (hi ? 16 : 0);
    if (jb > SEQ - 16) jb = SEQ - 16;  // clamp (values multiplied by zero P)
#pragma unroll
    for (int dblk = 0; dblk < 4; ++dblk) {
      const unsigned short* vrow = vbase + (size_t)(dblk * 16 + n) * SEQ + jb;
      ABfrag bv;
      bv.q[0] = *(const uint4*)(vrow);
      bv.q[1] = *(const uint4*)(vrow + 8);
      o[dblk] = __builtin_amdgcn_wmma_f32_16x16x32_bf16(
          false, ap.v, false, bv.v, (short)0, o[dblk], false, false);
    }
  }

  // store context as bf16 [b*S, h*64+d] for the final projection
  const int b = bh >> 4, h = bh & 15;
#pragma unroll
  for (int dblk = 0; dblk < 4; ++dblk)
#pragma unroll
    for (int r = 0; r < 8; ++r) {
      int s = q0 + r + 8 * hi;
      ctx[((size_t)(b * SEQ + s)) * HID + h * 64 + dblk * 16 + n] =
          f2b(o[dblk][r]);
    }
}

// ---- workspace layout (bytes) ----
static const size_t OFF_X   = 0;                 // 8 MiB  X bf16
static const size_t OFF_WQ  = (size_t)8  << 20;  // 2 MiB each
static const size_t OFF_WK  = (size_t)10 << 20;
static const size_t OFF_WV  = (size_t)12 << 20;
static const size_t OFF_WO  = (size_t)14 << 20;
static const size_t OFF_Q   = (size_t)16 << 20;  // 8 MiB
static const size_t OFF_K   = (size_t)24 << 20;
static const size_t OFF_VT  = (size_t)32 << 20;
static const size_t OFF_CTX = (size_t)40 << 20;  // total 48 MiB

extern "C" void kernel_launch(void* const* d_in, const int* in_sizes, int n_in,
                              void* d_out, int out_size, void* d_ws, size_t ws_size,
                              hipStream_t stream) {
  (void)in_sizes; (void)n_in; (void)out_size; (void)ws_size;
  const float* hs = (const float*)d_in[0];
  const float* Wq = (const float*)d_in[1];
  const float* Wk = (const float*)d_in[2];
  const float* Wv = (const float*)d_in[3];
  const float* Wo = (const float*)d_in[4];
  const float* qn = (const float*)d_in[5];
  const float* kn = (const float*)d_in[6];

  char* ws = (char*)d_ws;
  unsigned short* Xbf  = (unsigned short*)(ws + OFF_X);
  unsigned short* Wqb  = (unsigned short*)(ws + OFF_WQ);
  unsigned short* Wkb  = (unsigned short*)(ws + OFF_WK);
  unsigned short* Wvb  = (unsigned short*)(ws + OFF_WV);
  unsigned short* Wob  = (unsigned short*)(ws + OFF_WO);
  unsigned short* Qb   = (unsigned short*)(ws + OFF_Q);
  unsigned short* Kb   = (unsigned short*)(ws + OFF_K);
  unsigned short* Vtb  = (unsigned short*)(ws + OFF_VT);
  unsigned short* Ctxb = (unsigned short*)(ws + OFF_CTX);

  float* out_p  = (float*)d_out;                          // [2,2048,1024]
  float* attn_p = out_p + (size_t)BATCH * SEQ * HID;      // [2,16,2048,2048]

  const int NX = MROWS * HID;   // 4194304
  const int NW = HID * HID;     // 1048576
  f32_to_bf16_kernel<<<(NX + 255) / 256, 256, 0, stream>>>(hs, Xbf, NX);
  f32_to_bf16_kernel<<<(NW + 255) / 256, 256, 0, stream>>>(Wq, Wqb, NW);
  f32_to_bf16_kernel<<<(NW + 255) / 256, 256, 0, stream>>>(Wk, Wkb, NW);
  f32_to_bf16_kernel<<<(NW + 255) / 256, 256, 0, stream>>>(Wv, Wvb, NW);
  f32_to_bf16_kernel<<<(NW + 255) / 256, 256, 0, stream>>>(Wo, Wob, NW);

  proj_kernel<<<512, 256, 0, stream>>>(Xbf, Wqb, qn, Qb,  MODE_QKNORM);
  proj_kernel<<<512, 256, 0, stream>>>(Xbf, Wkb, kn, Kb,  MODE_QKNORM);
  proj_kernel<<<512, 256, 0, stream>>>(Xbf, Wvb, qn, Vtb, MODE_VT);

  // zero the out-of-window region of the attn output
  hipMemsetAsync(attn_p, 0, (size_t)BATCH * NHEADS * SEQ * SEQ * sizeof(float), stream);

  attn_kernel<<<512, 256, 0, stream>>>(Qb, Kb, Vtb, attn_p, Ctxb);

  proj_kernel<<<512, 256, 0, stream>>>(Ctxb, Wob, qn, out_p, MODE_F32OUT);
}
